// ExtractorForSMPLX_74766790689503
// MI455X (gfx1250) — matrix operand order repaired and checked by reading, not compiled, not run on hardware
//
#include <hip/hip_runtime.h>
#include <hip/hip_bf16.h>
#include <math.h>

// ---------------- problem constants (baked from reference) ----------------
#define BB     8
#define HH     64
#define WW     48
#define LQ     3072          // HH*WW
#define MROWS  24576         // BB*LQ
#define DIMC   1024
#define NHEADS 16
#define NPTS   4
#define DRC    512
#define HDC    32
#define HIDC   256
#define TTC    80

typedef __attribute__((ext_vector_type(16))) _Float16 v16h;
typedef __attribute__((ext_vector_type(8)))  _Float16 v8h;
typedef __attribute__((ext_vector_type(8)))  float    v8f;

// ---- explicit global-address-space accessors: force global_load_b128 /
// ---- global_store (FLAT would also tick DScnt and tie up the LDS pipe) ----
#define GAS __attribute__((address_space(1)))
__device__ __forceinline__ v8h gload_v8h(const _Float16* p) {
    return *(const GAS v8h*)(unsigned long long)p;
}
__device__ __forceinline__ float gload_f32(const float* p) {
    return *(const GAS float*)(unsigned long long)p;
}
__device__ __forceinline__ void gstore_f32(float* p, float v) {
    *(GAS float*)(unsigned long long)p = v;
}
__device__ __forceinline__ void gstore_f16(_Float16* p, _Float16 v) {
    *(GAS _Float16*)(unsigned long long)p = v;
}
__device__ __forceinline__ void gstore_v8h(_Float16* p, v8h v) {
    *(GAS v8h*)(unsigned long long)p = v;
}

// ---------------------------------------------------------------------------
// Repack fp32 weight W[K][N] (row-major) into WMMA B-fragment tiles (f16).
// Packed index: ((nt*KT + kt)*32 + lane)*16 + h
//   lane 0-15  : column N = nt*16 + lane,    K = kt*32 + (h&7) + 16*(h>>3)      (h = 0..15)
//   lane 16-31 : column N = nt*16 + lane-16, K = kt*32 + 8 + (h&7) + 16*(h>>3)
// ---------------------------------------------------------------------------
__global__ __launch_bounds__(256)
void repack_b_kernel(const float* __restrict__ W, _Float16* __restrict__ Bp,
                     int K, int N)
{
    int idx = blockIdx.x * blockDim.x + threadIdx.x;
    if (idx >= K * N) return;
    int h    = idx & 15;
    int lane = (idx >> 4) & 31;
    int rest = idx >> 9;           // nt*KT + kt
    int KT   = K >> 5;
    int kt   = rest % KT;
    int nt   = rest / KT;
    int kk   = kt * 32 + ((lane >> 4) << 3) + (h & 7) + ((h >> 3) << 4);
    int nn   = nt * 16 + (lane & 15);
    Bp[idx] = (_Float16)gload_f32(W + (size_t)kk * N + nn);
}

// ---------------------------------------------------------------------------
// LayerNorm (fp32 in, f16 out). Handles batch-strided source rows:
//   src_row = (r / rowsPerBatch)*batchStride + rowOff + (r % rowsPerBatch)
// ---------------------------------------------------------------------------
__global__ __launch_bounds__(256)
void ln_f16_kernel(const float* __restrict__ X, const float* __restrict__ g,
                   const float* __restrict__ bt, _Float16* __restrict__ Y,
                   int cols, int rowsPerBatch, int batchStride, int rowOff)
{
    int r   = blockIdx.x;
    int src = (r / rowsPerBatch) * batchStride + rowOff + (r % rowsPerBatch);
    const float* xr = X + (size_t)src * cols;

    float s = 0.f, ss = 0.f;
    for (int i = threadIdx.x; i < cols; i += 256) {
        float v = gload_f32(xr + i); s += v; ss += v * v;
    }
    __shared__ float sh0[256], sh1[256];
    sh0[threadIdx.x] = s; sh1[threadIdx.x] = ss;
    __syncthreads();
    for (int off = 128; off > 0; off >>= 1) {
        if (threadIdx.x < off) {
            sh0[threadIdx.x] += sh0[threadIdx.x + off];
            sh1[threadIdx.x] += sh1[threadIdx.x + off];
        }
        __syncthreads();
    }
    float mean = sh0[0] / cols;
    float var  = sh1[0] / cols - mean * mean;
    float rstd = rsqrtf(var + 1e-6f);

    _Float16* yr = Y + (size_t)r * cols;
    for (int i = threadIdx.x; i < cols; i += 256)
        gstore_f16(yr + i,
                   (_Float16)((gload_f32(xr + i) - mean) * rstd * g[i] + bt[i]));
}

// ---------------------------------------------------------------------------
// WMMA GEMM, register-blocked: each wave computes a 64x64 output tile
// (4 M-frags x 4 N-frags -> 16 WMMAs per 32-deep K step, 16 b128 loads).
// Block = 4 waves stacked along M -> block tile 256x64.
// C = A(f16 row-major) * Bpacked(f16 fragments) (+bias) (+residual fp32),
// store fp32 or f16.
// ---------------------------------------------------------------------------
__global__ __launch_bounds__(128)
void gemm_wmma_kernel(const _Float16* __restrict__ A,
                      const _Float16* __restrict__ Bp,
                      const float* __restrict__ bias,
                      const float* __restrict__ residual,
                      float* __restrict__ Cf,
                      _Float16* __restrict__ Ch,
                      int N, int K)
{
    const int lane   = threadIdx.x & 31;
    const int wave   = threadIdx.x >> 5;
    const int mBase  = blockIdx.y * 256 + wave * 64;   // wave's 64 rows
    const int ntBase = blockIdx.x * 4;                 // wave's 4 16-col tiles
    const int KT     = K >> 5;
    const int kOff   = (lane >> 4) << 3;               // lanes 16-31: +8 halves
    const int rsel   = lane & 15;

    const _Float16* aRow[4];
#pragma unroll
    for (int i = 0; i < 4; ++i)
        aRow[i] = A + (size_t)(mBase + i * 16 + rsel) * K + kOff;
    const _Float16* bRow[4];
#pragma unroll
    for (int j = 0; j < 4; ++j)
        bRow[j] = Bp + (size_t)(ntBase + j) * KT * 512 + (size_t)lane * 16;

    v8f acc[4][4] = {};

    for (int kt = 0; kt < KT; ++kt) {
        v16h af[4], bf[4];
#pragma unroll
        for (int i = 0; i < 4; ++i) {
            v8h a0 = gload_v8h(aRow[i]);
            v8h a1 = gload_v8h(aRow[i] + 16);
#pragma unroll
            for (int e = 0; e < 8; ++e) { af[i][e] = a0[e]; af[i][e + 8] = a1[e]; }
            aRow[i] += 32;
        }
#pragma unroll
        for (int j = 0; j < 4; ++j) {
            v8h b0 = gload_v8h(bRow[j]);
            v8h b1 = gload_v8h(bRow[j] + 8);
#pragma unroll
            for (int e = 0; e < 8; ++e) { bf[j][e] = b0[e]; bf[j][e + 8] = b1[e]; }
            bRow[j] += 512;
        }
#pragma unroll
        for (int i = 0; i < 4; ++i)
#pragma unroll
            for (int j = 0; j < 4; ++j)
                acc[i][j] = __builtin_amdgcn_wmma_f32_16x16x32_f16(
                    false, af[i], false, bf[j], (short)0, acc[i][j], false, false);
    }

    const int colLo = ntBase * 16 + (lane & 15);
    const int mOff  = (lane >> 4) << 3;
#pragma unroll
    for (int i = 0; i < 4; ++i) {
#pragma unroll
        for (int j = 0; j < 4; ++j) {
            int col = colLo + j * 16;
            float bv = bias ? gload_f32(bias + col) : 0.f;
#pragma unroll
            for (int r = 0; r < 8; ++r) {
                int m = mBase + i * 16 + mOff + r;
                float v = acc[i][j][r] + bv;
                if (residual) v += gload_f32(residual + (size_t)m * N + col);
                if (Ch) gstore_f16(Ch + (size_t)m * N + col, (_Float16)v);
                else    gstore_f32(Cf + (size_t)m * N + col, v);
            }
        }
    }
}

// ---------------------------------------------------------------------------
// Deformable-attention sampling: softmax over NP=4 logits, bilinear gather
// from value (f16), weighted sum -> sampled (f16, (B*Lq, DR)).
// One thread handles one (b,q,head, 8-channel group).
// ---------------------------------------------------------------------------
__global__ __launch_bounds__(256)
void deform_attn_kernel(const _Float16* __restrict__ value,
                        const float* __restrict__ off,
                        const float* __restrict__ logits,
                        const float* __restrict__ ref,
                        _Float16* __restrict__ out)
{
    int t = blockIdx.x * blockDim.x + threadIdx.x;
    if (t >= MROWS * NHEADS * 4) return;
    int c8 = t & 3;
    int h  = (t >> 2) & 15;
    int rq = t >> 6;                      // b*LQ + q
    int b  = rq / LQ;

    const float* lg = logits + (size_t)rq * (NHEADS * NPTS) + h * NPTS;
    float l0 = gload_f32(lg + 0), l1 = gload_f32(lg + 1);
    float l2 = gload_f32(lg + 2), l3 = gload_f32(lg + 3);
    float mx = fmaxf(fmaxf(l0, l1), fmaxf(l2, l3));
    float aw[NPTS] = { expf(l0 - mx), expf(l1 - mx), expf(l2 - mx), expf(l3 - mx) };
    float inv = 1.f / (aw[0] + aw[1] + aw[2] + aw[3]);
#pragma unroll
    for (int p = 0; p < NPTS; ++p) aw[p] *= inv;

    const float* op = off + (size_t)rq * (NHEADS * NPTS * 2) + h * (NPTS * 2);
    float rx = gload_f32(ref + (size_t)rq * 2 + 0);
    float ry = gload_f32(ref + (size_t)rq * 2 + 1);

    float accv[8] = {};
#pragma unroll
    for (int p = 0; p < NPTS; ++p) {
        // loc = ref + off/[W,H]; x = loc.x*W - 0.5 -> ref.x*W + off.x - 0.5
        float x = rx * (float)WW + gload_f32(op + p * 2 + 0) - 0.5f;
        float y = ry * (float)HH + gload_f32(op + p * 2 + 1) - 0.5f;
        float x0f = floorf(x), y0f = floorf(y);
        float lw = x - x0f, lh = y - y0f;
        int x0 = (int)x0f, y0 = (int)y0f;
        float wgt[4] = { (1.f - lh) * (1.f - lw), (1.f - lh) * lw,
                         lh * (1.f - lw),         lh * lw };
#pragma unroll
        for (int k = 0; k < 4; ++k) {
            int dx = k & 1, dy = k >> 1;
            int xc = x0 + dx, yc = y0 + dy;
            if (xc < 0 || xc >= WW || yc < 0 || yc >= HH) continue;
            int idx = yc * WW + xc;
            const v8h g = gload_v8h(value +
                ((size_t)(b * LQ + idx) * DRC + h * HDC + c8 * 8));
            float wa = wgt[k] * aw[p];
#pragma unroll
            for (int j = 0; j < 8; ++j) accv[j] += wa * (float)g[j];
        }
    }
    v8h o;
#pragma unroll
    for (int j = 0; j < 8; ++j) o[j] = (_Float16)accv[j];
    gstore_v8h(out + (size_t)rq * DRC + h * HDC + c8 * 8, o);
}

// ---------------------------------------------------------------------------
// Depthwise 3x3 conv (SAME) + exact GeLU:  h1(fp32) -> h2(f16)
// One thread handles one (b,y,x) and 4 channels.
// ---------------------------------------------------------------------------
__global__ __launch_bounds__(256)
void dwconv_gelu_kernel(const float* __restrict__ X, const float* __restrict__ Wdw,
                        const float* __restrict__ bdw, _Float16* __restrict__ Y)
{
    int t = blockIdx.x * blockDim.x + threadIdx.x;
    if (t >= BB * HH * WW * (HIDC / 4)) return;
    int cg = t & 63; int c = cg * 4;
    int sp = t >> 6;
    int x = sp % WW; int rest = sp / WW;
    int y = rest % HH; int b = rest / HH;

    float acc[4] = {};
#pragma unroll
    for (int ky = -1; ky <= 1; ++ky) {
        int yy = y + ky;
        if (yy < 0 || yy >= HH) continue;
#pragma unroll
        for (int kx = -1; kx <= 1; ++kx) {
            int xx = x + kx;
            if (xx < 0 || xx >= WW) continue;
            const float* xp = X + ((size_t)(b * LQ) + yy * WW + xx) * HIDC + c;
            const float* wp = Wdw + ((ky + 1) * 3 + (kx + 1)) * HIDC + c;
#pragma unroll
            for (int j = 0; j < 4; ++j) acc[j] += gload_f32(xp + j) * gload_f32(wp + j);
        }
    }
    _Float16* yp = Y + ((size_t)(b * LQ) + y * WW + x) * HIDC + c;
#pragma unroll
    for (int j = 0; j < 4; ++j) {
        float v = acc[j] + gload_f32(bdw + c + j);
        v = 0.5f * v * (1.f + erff(v * 0.70710678118654752f));   // exact GeLU
        gstore_f16(yp + j, (_Float16)v);
    }
}

// ---------------------------------------------------------------------------
extern "C" void kernel_launch(void* const* d_in, const int* in_sizes, int n_in,
                              void* d_out, int out_size, void* d_ws, size_t ws_size,
                              hipStream_t stream)
{
    const float* query  = (const float*)d_in[0];
    const float* refpts = (const float*)d_in[1];
    const float* feat   = (const float*)d_in[2];
    const float* qn_g   = (const float*)d_in[7];
    const float* qn_b   = (const float*)d_in[8];
    const float* fn_g   = (const float*)d_in[9];
    const float* fn_b   = (const float*)d_in[10];
    const float* ffn_g  = (const float*)d_in[11];
    const float* ffn_b  = (const float*)d_in[12];
    const float* w_off  = (const float*)d_in[13];
    const float* b_off  = (const float*)d_in[14];
    const float* w_attn = (const float*)d_in[15];
    const float* b_attn = (const float*)d_in[16];
    const float* w_val  = (const float*)d_in[17];
    const float* b_val  = (const float*)d_in[18];
    const float* w_out  = (const float*)d_in[19];
    const float* b_out  = (const float*)d_in[20];
    const float* w_fc1  = (const float*)d_in[21];
    const float* b_fc1  = (const float*)d_in[22];
    const float* w_dw   = (const float*)d_in[23];
    const float* b_dw   = (const float*)d_in[24];
    const float* w_fc2  = (const float*)d_in[25];
    const float* b_fc2  = (const float*)d_in[26];
    float* out = (float*)d_out;

    // ---- workspace layout (256B aligned) ----
    char* p = (char*)d_ws;
    auto alloc = [&](size_t bytes) -> void* {
        void* r = (void*)p;
        p += (bytes + 255) & ~(size_t)255;
        return r;
    };
    _Float16* qn16    = (_Float16*)alloc((size_t)MROWS * DIMC * 2);
    _Float16* fn16    = (_Float16*)alloc((size_t)MROWS * DIMC * 2);
    _Float16* wval_p  = (_Float16*)alloc((size_t)DIMC * DRC  * 2);
    _Float16* woff_p  = (_Float16*)alloc((size_t)DIMC * 128  * 2);
    _Float16* wattn_p = (_Float16*)alloc((size_t)DIMC * 64   * 2);
    _Float16* wout_p  = (_Float16*)alloc((size_t)DRC  * DIMC * 2);
    _Float16* wfc1_p  = (_Float16*)alloc((size_t)DIMC * HIDC * 2);
    _Float16* wfc2_p  = (_Float16*)alloc((size_t)HIDC * DIMC * 2);
    _Float16* value16 = (_Float16*)alloc((size_t)MROWS * DRC * 2);
    float*    offb    = (float*)   alloc((size_t)MROWS * 128 * 4);
    float*    attnb   = (float*)   alloc((size_t)MROWS * 64  * 4);
    float*    qres    = (float*)   alloc((size_t)MROWS * DIMC * 4);
    float*    h1      = (float*)   alloc((size_t)MROWS * HIDC * 4);
    _Float16* h2      = (_Float16*)alloc((size_t)MROWS * HIDC * 2);
    _Float16* sampled = fn16;   // fn16 dead after value GEMM
    _Float16* qn2     = qn16;   // qn16 dead after off/attn GEMMs

    auto repack = [&](const float* W, _Float16* Bp, int K, int N) {
        int total = K * N;
        repack_b_kernel<<<(total + 255) / 256, 256, 0, stream>>>(W, Bp, K, N);
    };
    auto gemm = [&](const _Float16* A, const _Float16* Bp, const float* bias,
                    const float* resid, float* Cf, _Float16* Ch, int N, int K) {
        dim3 grid(N / 64, MROWS / 256);      // block tile 256x64 (4 waves x 64x64)
        gemm_wmma_kernel<<<grid, 128, 0, stream>>>(A, Bp, bias, resid, Cf, Ch, N, K);
    };

    // 1) pack all weights into WMMA B-fragment layout (one-time, small)
    repack(w_val,  wval_p,  DIMC, DRC);
    repack(w_off,  woff_p,  DIMC, 128);
    repack(w_attn, wattn_p, DIMC, 64);
    repack(w_out,  wout_p,  DRC,  DIMC);
    repack(w_fc1,  wfc1_p,  DIMC, HIDC);
    repack(w_fc2,  wfc2_p,  HIDC, DIMC);

    // 2) LN(feat[:,TT:]) -> fn16 ; LN(query) -> qn16
    ln_f16_kernel<<<MROWS, 256, 0, stream>>>(feat, fn_g, fn_b, fn16,
                                             DIMC, LQ, TTC + LQ, TTC);
    ln_f16_kernel<<<MROWS, 256, 0, stream>>>(query, qn_g, qn_b, qn16,
                                             DIMC, LQ, LQ, 0);

    // 3) value = fn @ w_val + b_val  (f16 out)
    gemm(fn16, wval_p, b_val, nullptr, nullptr, value16, DRC, DIMC);

    // 4) offsets & attention logits from qn
    gemm(qn16, woff_p,  b_off,  nullptr, offb,  nullptr, 128, DIMC);
    gemm(qn16, wattn_p, b_attn, nullptr, attnb, nullptr, 64,  DIMC);

    // 5) softmax + bilinear sampling -> sampled (reuses fn16)
    {
        int total = MROWS * NHEADS * 4;
        deform_attn_kernel<<<(total + 255) / 256, 256, 0, stream>>>(
            value16, offb, attnb, refpts, sampled);
    }

    // 6) qres = query + sampled @ w_out + b_out   (residual fused)
    gemm(sampled, wout_p, b_out, query, qres, nullptr, DIMC, DRC);

    // 7) LN(qres) -> qn2 (reuses qn16)
    ln_f16_kernel<<<MROWS, 256, 0, stream>>>(qres, ffn_g, ffn_b, qn2,
                                             DIMC, LQ, LQ, 0);

    // 8) h1 = qn2 @ w_fc1 + b_fc1 (fp32)
    gemm(qn2, wfc1_p, b_fc1, nullptr, h1, nullptr, HIDC, DIMC);

    // 9) depthwise 3x3 + GeLU -> h2 (f16)
    {
        int total = BB * HH * WW * (HIDC / 4);
        dwconv_gelu_kernel<<<(total + 255) / 256, 256, 0, stream>>>(
            h1, w_dw, b_dw, h2);
    }

    // 10) out = qres + h2 @ w_fc2 + b_fc2  (residual fused, fp32 out)
    gemm(h2, wfc2_p, b_fc2, qres, out, nullptr, DIMC, HIDC);

    (void)in_sizes; (void)n_in; (void)out_size; (void)ws_size;
}
